// GATFactVerifier_82918638617114
// MI455X (gfx1250) — matrix operand order repaired and checked by reading, not compiled, not run on hardware
//
#include <hip/hip_runtime.h>
#include <hip/hip_bf16.h>
#include <math.h>

// Problem constants (match reference)
#define NND  20000     // nodes
#define NE   100000    // edges
#define NEP  120000    // edges + self loops
#define NG   128       // graphs
#define DIN  768
#define DHID 512
#define NH   8
#define CH   64        // per-head channels for concat layers
#define DOUT 3
#define NEG_SLOPE 0.2f

typedef __attribute__((ext_vector_type(16))) __bf16 v16bf;
typedef __attribute__((ext_vector_type(8)))  float  v8f;
typedef __attribute__((ext_vector_type(4)))  unsigned int u32x4;

union BF16Frag {
  v16bf v;
  unsigned int u[8];
  unsigned short s[16];
};

static __device__ __forceinline__ unsigned short f2bf(float f) {
  unsigned int u = __float_as_uint(f);
  u += 0x7FFFu + ((u >> 16) & 1u);   // round-to-nearest-even
  return (unsigned short)(u >> 16);
}

static __device__ __forceinline__ void atomicMaxFloat(float* addr, float val) {
  // valid when addr initialized to -inf (0xFF800000)
  if (val >= 0.0f) atomicMax((int*)addr, __float_as_int(val));
  else             atomicMin((unsigned int*)addr, __float_as_uint(val));
}

// ---------------------------------------------------------------- utilities
__global__ void gatk_fill(float* __restrict__ p, float v, int n) {
  int i = blockIdx.x * blockDim.x + threadIdx.x;
  if (i < n) p[i] = v;
}

__global__ void gatk_f32_to_bf16(const float* __restrict__ in,
                                 unsigned short* __restrict__ out, int n) {
  int i = blockIdx.x * blockDim.x + threadIdx.x;
  if (i < n) out[i] = f2bf(in[i]);
}

// W[k, n] (K x Ntot row-major, f32) -> Wt[n, k] (Ntot x K row-major, bf16)
// Makes the WMMA B-fragment a contiguous K-run per lane (2x global_load_b128).
__global__ void gatk_w_to_bf16t(const float* __restrict__ W,
                                unsigned short* __restrict__ Wt,
                                int Kdim, int Ntot) {
  long long idx = (long long)blockIdx.x * blockDim.x + threadIdx.x;
  if (idx >= (long long)Kdim * Ntot) return;
  int n = (int)(idx / Kdim);
  int k = (int)(idx % Kdim);
  Wt[idx] = f2bf(W[(size_t)k * Ntot + n]);
}

__global__ void gatk_build_edges(const long long* __restrict__ ei,
                                 int* __restrict__ src, int* __restrict__ dst) {
  int i = blockIdx.x * blockDim.x + threadIdx.x;
  if (i >= NEP) return;
  if (i < NE) { src[i] = (int)ei[i]; dst[i] = (int)ei[NE + i]; }
  else        { src[i] = i - NE;     dst[i] = i - NE; }        // self loops
}

// ---------------------------------------------------------------- WMMA GEMM
// C[M x Nc] = A[M x K] * Bt[bcol0+n, k]^T  (A row-major bf16, Bt = W^T bf16)
// One wave computes a 16x64 strip: 4 accumulators, 4 WMMAs per 32-wide K step.
__global__ __launch_bounds__(256)
void gatk_wmma_gemm(const unsigned short* __restrict__ A,
                    const unsigned short* __restrict__ Bt,
                    float* __restrict__ C,
                    int M, int Nc, int K, int bcol0) {
  const int lane = threadIdx.x & 31;
  const int wave = threadIdx.x >> 5;
  const int tile = blockIdx.x * 8 + wave;          // wave-uniform
  if (tile * 16 >= M) return;                      // whole wave exits together
  const int m0   = tile * 16;
  const int n0   = blockIdx.y * 64;
  const int half = lane >> 4;                      // 0: lanes 0-15, 1: lanes 16-31
  const int l15  = lane & 15;

  const unsigned short* arow = A + (size_t)(m0 + l15) * K;
  const unsigned short* br0 = Bt + (size_t)(bcol0 + n0 +  0 + l15) * K;
  const unsigned short* br1 = Bt + (size_t)(bcol0 + n0 + 16 + l15) * K;
  const unsigned short* br2 = Bt + (size_t)(bcol0 + n0 + 32 + l15) * K;
  const unsigned short* br3 = Bt + (size_t)(bcol0 + n0 + 48 + l15) * K;

  v8f acc0 = {}, acc1 = {}, acc2 = {}, acc3 = {};

  for (int kk = 0; kk < K; kk += 32) {
    BF16Frag a, b0, b1, b2, b3;
    // A 16x32 fragment: lanes 0-15 hold K=0..15, lanes 16-31 K=16..31
    *(u32x4*)&a.u[0] = *(const u32x4*)(arow + kk + (half << 4));
    *(u32x4*)&a.u[4] = *(const u32x4*)(arow + kk + 8 + (half << 4));
    // B 32x16 fragment: lanes 0-15 hold K=0..15, lanes 16-31 K=16..31 (contig in Wt)
    const int kb = kk + (half << 4);
    *(u32x4*)&b0.u[0] = *(const u32x4*)(br0 + kb);
    *(u32x4*)&b0.u[4] = *(const u32x4*)(br0 + kb + 8);
    *(u32x4*)&b1.u[0] = *(const u32x4*)(br1 + kb);
    *(u32x4*)&b1.u[4] = *(const u32x4*)(br1 + kb + 8);
    *(u32x4*)&b2.u[0] = *(const u32x4*)(br2 + kb);
    *(u32x4*)&b2.u[4] = *(const u32x4*)(br2 + kb + 8);
    *(u32x4*)&b3.u[0] = *(const u32x4*)(br3 + kb);
    *(u32x4*)&b3.u[4] = *(const u32x4*)(br3 + kb + 8);
    if (kk + 32 < K) __builtin_prefetch(arow + kk + 32, 0, 0);
    acc0 = __builtin_amdgcn_wmma_f32_16x16x32_bf16(false, a.v, false, b0.v, (short)0, acc0, false, false);
    acc1 = __builtin_amdgcn_wmma_f32_16x16x32_bf16(false, a.v, false, b1.v, (short)0, acc1, false, false);
    acc2 = __builtin_amdgcn_wmma_f32_16x16x32_bf16(false, a.v, false, b2.v, (short)0, acc2, false, false);
    acc3 = __builtin_amdgcn_wmma_f32_16x16x32_bf16(false, a.v, false, b3.v, (short)0, acc3, false, false);
  }
  // C/D layout: lane = N%16 (+16 for M>=8), VGPR r -> M = r (+8)
#pragma unroll
  for (int r = 0; r < 8; ++r) {
    float* crow = C + (size_t)(m0 + (half << 3) + r) * Nc + n0 + l15;
    crow[0]  = acc0[r];
    crow[16] = acc1[r];
    crow[32] = acc2[r];
    crow[48] = acc3[r];
  }
}

// ---------------------------------------------------------------- attention
// e_src[n,h] = <hw[n,h,:], a_src[h,:]>, likewise e_dst
__global__ void gatk_attn_scores(const float* __restrict__ hw,
                                 const float* __restrict__ as,
                                 const float* __restrict__ ad,
                                 float* __restrict__ es, float* __restrict__ ed,
                                 int H, int C) {
  int idx = blockIdx.x * blockDim.x + threadIdx.x;
  if (idx >= NND * H) return;
  int n = idx / H, h = idx % H;
  const float* row = hw + (size_t)n * H * C + (size_t)h * C;
  float s = 0.f, d = 0.f;
  for (int c = 0; c < C; ++c) {
    float v = row[c];
    s += v * as[h * C + c];
    d += v * ad[h * C + c];
  }
  es[idx] = s;
  ed[idx] = d;
}

__global__ void gatk_edge_max(const int* __restrict__ src, const int* __restrict__ dst,
                              const float* __restrict__ es, const float* __restrict__ ed,
                              float* __restrict__ m, int H) {
  int idx = blockIdx.x * blockDim.x + threadIdx.x;
  if (idx >= NEP * H) return;
  int e = idx / H, h = idx % H;
  int s = src[e], d = dst[e];
  float v = es[s * H + h] + ed[d * H + h];
  v = (v > 0.f) ? v : NEG_SLOPE * v;               // leaky relu
  atomicMaxFloat(&m[d * H + h], v);
}

__global__ void gatk_edge_exp_sum(const int* __restrict__ src, const int* __restrict__ dst,
                                  const float* __restrict__ es, const float* __restrict__ ed,
                                  const float* __restrict__ m, float* __restrict__ aeg,
                                  float* __restrict__ den, int H) {
  int idx = blockIdx.x * blockDim.x + threadIdx.x;
  if (idx >= NEP * H) return;
  int e = idx / H, h = idx % H;
  int s = src[e], d = dst[e];
  float v = es[s * H + h] + ed[d * H + h];
  v = (v > 0.f) ? v : NEG_SLOPE * v;
  float a = __expf(v - m[d * H + h]);
  aeg[idx] = a;
  atomicAdd(&den[d * H + h], a);
}

// msg[dst, j..j+3] += a[e, h(j)] * hw[src, j..j+3]; H*C == 512 in all layers
__global__ void gatk_edge_msg4(const int* __restrict__ src, const int* __restrict__ dst,
                               const float* __restrict__ aeg, const float* __restrict__ hw,
                               float* __restrict__ msg, int H, int C) {
  long long idx = (long long)blockIdx.x * blockDim.x + threadIdx.x;
  if (idx >= (long long)NEP * (DHID / 4)) return;
  int e = (int)(idx / (DHID / 4));
  int j = (int)(idx % (DHID / 4)) * 4;
  int h = j / C;                                   // C is a multiple of 4
  int s = src[e], d = dst[e];
  float a = aeg[e * H + h];
  const float4 hv = *(const float4*)&hw[(size_t)s * DHID + j];
  float* mo = &msg[(size_t)d * DHID + j];
  atomicAdd(mo + 0, a * hv.x);
  atomicAdd(mo + 1, a * hv.y);
  atomicAdd(mo + 2, a * hv.z);
  atomicAdd(mo + 3, a * hv.w);
}

// out[n,j] = msg[n,j]/max(den[n,h],eps) + bias[j]  (+relu for hidden layers)
__global__ void gatk_finalize(const float* __restrict__ msg, const float* __restrict__ den,
                              const float* __restrict__ bias, float* __restrict__ out,
                              int H, int C, int do_relu) {
  long long idx = (long long)blockIdx.x * blockDim.x + threadIdx.x;
  if (idx >= (long long)NND * DHID) return;
  int n = (int)(idx / DHID);
  int j = (int)(idx % DHID);
  int h = j / C;
  float v = msg[idx] / fmaxf(den[n * H + h], 1e-16f) + bias[j];
  if (do_relu) v = fmaxf(v, 0.f);
  out[idx] = v;
}

// layer-5 head accumulation: acc += (msg/den) / 8
__global__ void gatk_head_accum(const float* __restrict__ msg, const float* __restrict__ den,
                                float* __restrict__ acc) {
  long long idx = (long long)blockIdx.x * blockDim.x + threadIdx.x;
  if (idx >= (long long)NND * DHID) return;
  int n = (int)(idx / DHID);
  acc[idx] += 0.125f * msg[idx] / fmaxf(den[n], 1e-16f);
}

__global__ void gatk_add_bias(float* __restrict__ h, const float* __restrict__ b) {
  long long idx = (long long)blockIdx.x * blockDim.x + threadIdx.x;
  if (idx >= (long long)NND * DHID) return;
  h[idx] += b[idx % DHID];
}

// ---------------------------------------------------------------- pooling + classifier
__global__ void gatk_pool_count(const long long* __restrict__ batch, float* __restrict__ counts) {
  int i = blockIdx.x * blockDim.x + threadIdx.x;
  if (i < NND) atomicAdd(&counts[(int)batch[i]], 1.0f);
}

__global__ void gatk_pool_sum(const long long* __restrict__ batch, const float* __restrict__ h,
                              float* __restrict__ pooled) {
  long long idx = (long long)blockIdx.x * blockDim.x + threadIdx.x;
  if (idx >= (long long)NND * DHID) return;
  int n = (int)(idx / DHID);
  int j = (int)(idx % DHID);
  atomicAdd(&pooled[(size_t)batch[n] * DHID + j], h[idx]);
}

__global__ void gatk_classify(const float* __restrict__ pooled, const float* __restrict__ counts,
                              const float* __restrict__ W, const float* __restrict__ b,
                              float* __restrict__ out) {
  int idx = blockIdx.x * blockDim.x + threadIdx.x;
  if (idx >= NG * DOUT) return;
  int g = idx / DOUT, o = idx % DOUT;
  float inv = 1.0f / fmaxf(counts[g], 1.0f);
  float s = b[o];
  const float* pg = pooled + (size_t)g * DHID;
  for (int j = 0; j < DHID; ++j) s += pg[j] * inv * W[j * DOUT + o];
  out[idx] = s;
}

// ---------------------------------------------------------------- host driver
extern "C" void kernel_launch(void* const* d_in, const int* in_sizes, int n_in,
                              void* d_out, int out_size, void* d_ws, size_t ws_size,
                              hipStream_t stream) {
  (void)in_sizes; (void)n_in; (void)out_size; (void)ws_size;

  const float*     x     = (const float*)d_in[0];
  const long long* ei    = (const long long*)d_in[1];
  const long long* batch = (const long long*)d_in[2];
  const float* Ws[6], *As[6], *Ad[6], *Bs[6];
  for (int l = 0; l < 6; ++l) {
    Ws[l] = (const float*)d_in[3 + l];
    As[l] = (const float*)d_in[9 + l];
    Ad[l] = (const float*)d_in[15 + l];
    Bs[l] = (const float*)d_in[21 + l];
  }
  const float* clsW = (const float*)d_in[27];
  const float* clsB = (const float*)d_in[28];
  float* out = (float*)d_out;

  // workspace carve-out
  char* p = (char*)d_ws;
  auto take = [&](size_t bytes) -> char* {
    char* q = p;
    p += (bytes + 255) & ~(size_t)255;
    return q;
  };
  int*            src32  = (int*)take((size_t)NEP * 4);
  int*            dst32  = (int*)take((size_t)NEP * 4);
  unsigned short* hb     = (unsigned short*)take((size_t)NND * DIN * 2);
  unsigned short* wb     = (unsigned short*)take((size_t)DHID * 4096 * 2);
  float*          hw     = (float*)take((size_t)NND * DHID * 4);
  float*          bufA   = (float*)take((size_t)NND * DHID * 4);
  float*          bufB   = (float*)take((size_t)NND * DHID * 4);
  float*          msg    = (float*)take((size_t)NND * DHID * 4);
  float*          es     = (float*)take((size_t)NND * NH * 4);
  float*          ed     = (float*)take((size_t)NND * NH * 4);
  float*          mmax   = (float*)take((size_t)NND * NH * 4);
  float*          den    = (float*)take((size_t)NND * NH * 4);
  float*          aeg    = (float*)take((size_t)NEP * NH * 4);
  float*          pooled = (float*)take((size_t)NG * DHID * 4);
  float*          counts = (float*)take((size_t)NG * 4);

  auto nb = [](long long n) { return dim3((unsigned)((n + 255) / 256)); };
  const dim3 gemm_grid((NND / 16 + 7) / 8, DHID / 64);  // 157 x 8 (16x64 strips per wave)

  gatk_build_edges<<<nb(NEP), 256, 0, stream>>>(ei, src32, dst32);

  // ---------------- layers 0..4 (concat, relu)
  for (int l = 0; l < 5; ++l) {
    const int K = (l == 0) ? DIN : DHID;
    const float* hin  = (l == 0) ? x : ((l & 1) ? bufA : bufB);
    float*       hout = (l & 1) ? bufB : bufA;

    gatk_f32_to_bf16<<<nb((long long)NND * K), 256, 0, stream>>>(hin, hb, NND * K);
    gatk_w_to_bf16t<<<nb((long long)K * DHID), 256, 0, stream>>>(Ws[l], wb, K, DHID);
    gatk_wmma_gemm<<<gemm_grid, 256, 0, stream>>>(hb, wb, hw, NND, DHID, K, 0);

    gatk_attn_scores<<<nb((long long)NND * NH), 256, 0, stream>>>(hw, As[l], Ad[l], es, ed, NH, CH);
    gatk_fill<<<nb((long long)NND * NH), 256, 0, stream>>>(mmax, -INFINITY, NND * NH);
    gatk_fill<<<nb((long long)NND * NH), 256, 0, stream>>>(den, 0.f, NND * NH);
    gatk_fill<<<nb((long long)NND * DHID), 256, 0, stream>>>(msg, 0.f, NND * DHID);

    gatk_edge_max<<<nb((long long)NEP * NH), 256, 0, stream>>>(src32, dst32, es, ed, mmax, NH);
    gatk_edge_exp_sum<<<nb((long long)NEP * NH), 256, 0, stream>>>(src32, dst32, es, ed, mmax, aeg, den, NH);
    gatk_edge_msg4<<<nb((long long)NEP * (DHID / 4)), 256, 0, stream>>>(src32, dst32, aeg, hw, msg, NH, CH);
    gatk_finalize<<<nb((long long)NND * DHID), 256, 0, stream>>>(msg, den, Bs[l], hout, NH, CH, 1);
  }

  // ---------------- layer 5 (8 heads of 512, averaged), input = bufA, acc = bufB
  gatk_f32_to_bf16<<<nb((long long)NND * DHID), 256, 0, stream>>>(bufA, hb, NND * DHID);
  gatk_w_to_bf16t<<<nb((long long)DHID * 4096), 256, 0, stream>>>(Ws[5], wb, DHID, 4096);
  gatk_fill<<<nb((long long)NND * DHID), 256, 0, stream>>>(bufB, 0.f, NND * DHID);

  for (int hh = 0; hh < NH; ++hh) {
    gatk_wmma_gemm<<<gemm_grid, 256, 0, stream>>>(hb, wb, hw, NND, DHID, DHID, hh * DHID);
    gatk_attn_scores<<<nb(NND), 256, 0, stream>>>(hw, As[5] + hh * DHID, Ad[5] + hh * DHID, es, ed, 1, DHID);
    gatk_fill<<<nb(NND), 256, 0, stream>>>(mmax, -INFINITY, NND);
    gatk_fill<<<nb(NND), 256, 0, stream>>>(den, 0.f, NND);
    gatk_fill<<<nb((long long)NND * DHID), 256, 0, stream>>>(msg, 0.f, NND * DHID);
    gatk_edge_max<<<nb(NEP), 256, 0, stream>>>(src32, dst32, es, ed, mmax, 1);
    gatk_edge_exp_sum<<<nb(NEP), 256, 0, stream>>>(src32, dst32, es, ed, mmax, aeg, den, 1);
    gatk_edge_msg4<<<nb((long long)NEP * (DHID / 4)), 256, 0, stream>>>(src32, dst32, aeg, hw, msg, 1, DHID);
    gatk_head_accum<<<nb((long long)NND * DHID), 256, 0, stream>>>(msg, den, bufB);
  }
  gatk_add_bias<<<nb((long long)NND * DHID), 256, 0, stream>>>(bufB, Bs[5]);

  // ---------------- pooling + classifier
  gatk_fill<<<nb((long long)NG * DHID), 256, 0, stream>>>(pooled, 0.f, NG * DHID);
  gatk_fill<<<nb(NG), 256, 0, stream>>>(counts, 0.f, NG);
  gatk_pool_count<<<nb(NND), 256, 0, stream>>>(batch, counts);
  gatk_pool_sum<<<nb((long long)NND * DHID), 256, 0, stream>>>(batch, bufB, pooled);
  gatk_classify<<<nb(NG * DOUT), 256, 0, stream>>>(pooled, counts, clsW, clsB, out);
}
